// PhysBiformer_42185168781832
// MI455X (gfx1250) — matrix operand order repaired and checked by reading, not compiled, not run on hardware
//
#include <hip/hip_runtime.h>
#include <hip/hip_bf16.h>

// ---------------------------------------------------------------------------
// PhysBiformer forward for MI455X (gfx1250, wave32).
// Heavy convs (stem1/stem2/patch) are implicit-GEMM on v_wmma_f32_16x16x32_f16
// with fused BN+ReLU+maxpool (or LIF) epilogues. stem1 stages its packed weight
// matrix into LDS with the Tensor Data Mover (tensor_load_to_lds, TENSORcnt).
// Token GEMMs share a WMMA kernel with fused bias/residual/LIF epilogues.
// ---------------------------------------------------------------------------

typedef _Float16 f16;
typedef __attribute__((ext_vector_type(16))) _Float16 v16h;
typedef __attribute__((ext_vector_type(8)))  _Float16 v8h;
typedef __attribute__((ext_vector_type(8)))  float    v8f;
typedef __attribute__((ext_vector_type(4)))  int      v4i;
typedef __attribute__((ext_vector_type(8)))  int      v8i;

// A-fragment K position for 16-bit WMMA A (16x32), contiguous order:
// lanes 0-15 / 16-31 hold K {0..15} / {16..31} in pairs.
__device__ __forceinline__ int a_kpos(int i, int half) {
  return half * 16 + 2 * i;
}

__device__ __forceinline__ unsigned wave_ballot(bool p) {
#if __has_builtin(__builtin_amdgcn_ballot_w32)
  return __builtin_amdgcn_ballot_w32(p);
#else
  return (unsigned)__ballot(p);
#endif
}

// Row-major A fragment: per lane = two contiguous 16B runs -> 2 vector loads.
__device__ __forceinline__ v16h load_a_frag(const f16* __restrict__ arow, int kb, int half) {
  v8h lo = *(const v8h*)(arow + kb + half * 16);
  v8h hi = *(const v8h*)(arow + kb + half * 16 + 8);
  return __builtin_shufflevector(lo, hi, 0, 1, 2, 3, 4, 5, 6, 7,
                                 8, 9, 10, 11, 12, 13, 14, 15);
}

// ------------------------------ weight packing ------------------------------

// conv weights already [N][Kreal] flat -> pad K to Kpad
__global__ void pack_pad(const float* __restrict__ w, f16* __restrict__ wt,
                         int Kreal, int Kpad, int total) {
  int id = blockIdx.x * blockDim.x + threadIdx.x;
  if (id >= total) return;
  int n = id / Kpad, k = id % Kpad;
  wt[id] = (k < Kreal) ? (f16)w[n * Kreal + k] : (f16)0.0f;
}

// linear weights are (K, N) -> transpose+pad into [N][Kpad]
__global__ void pack_T(const float* __restrict__ w, f16* __restrict__ wt,
                       int K, int N, int Kpad, int total) {
  int id = blockIdx.x * blockDim.x + threadIdx.x;
  if (id >= total) return;
  int n = id / Kpad, k = id % Kpad;
  wt[id] = (k < K) ? (f16)w[(long)k * N + n] : (f16)0.0f;
}

// ------------------------------ stem0 (direct) ------------------------------
// conv (16,3,1,5,5) pad(0,2,2) + BN + ReLU + maxpool(1,2,2), output f16
__global__ void stem0_fused(const float* __restrict__ x, const float* __restrict__ w,
                            const float* __restrict__ cb, const float* __restrict__ g,
                            const float* __restrict__ be, f16* __restrict__ out) {
  __shared__ float ws[1200];
  for (int i = threadIdx.x; i < 1200; i += blockDim.x) ws[i] = w[i];
  __syncthreads();
  long id = (long)blockIdx.x * blockDim.x + threadIdx.x;
  const long total = 2L * 16 * 160 * 64 * 64;
  if (id >= total) return;
  int w2 = (int)(id & 63);
  int h2 = (int)((id >> 6) & 63);
  int t  = (int)((id >> 12) % 160);
  int co = (int)((id / 655360) & 15);
  int b  = (int)(id / 10485760);
  float bias = cb[co], gg = g[co], bb = be[co];
  float m = 0.0f;
#pragma unroll
  for (int p = 0; p < 4; ++p) {
    int y = 2 * h2 + (p >> 1), xx = 2 * w2 + (p & 1);
    float s = 0.0f;
    for (int ci = 0; ci < 3; ++ci) {
      const float* ip = x + (((long)b * 3 + ci) * 160 + t) * 16384;
      const float* wp = ws + co * 75 + ci * 25;
#pragma unroll
      for (int kh = 0; kh < 5; ++kh) {
        int yi = y + kh - 2;
        if ((unsigned)yi >= 128u) continue;
#pragma unroll
        for (int kw = 0; kw < 5; ++kw) {
          int xi = xx + kw - 2;
          if ((unsigned)xi < 128u) s += ip[yi * 128 + xi] * wp[kh * 5 + kw];
        }
      }
    }
    float v = fmaxf(gg * (s + bias) + bb, 0.0f);
    m = fmaxf(m, v);
  }
  out[id] = (f16)m;
}

// ----------------- 3x3x3 conv (pad 1) + BN + ReLU + 2x2 pool ----------------
// Implicit GEMM: M = pre-pool voxels (tiled 16 = 4 pooled outputs x 4 pool pos),
// N = Cout (NT tiles of 16), K = Cin*27 padded. One wave per M-tile.
// NT==2 (stem1): weights staged to LDS via Tensor Data Mover.
template <int NT>
__global__ void conv_pool_wmma(const f16* __restrict__ in, const f16* __restrict__ wt,
                               const float* __restrict__ cb, const float* __restrict__ bg,
                               const float* __restrict__ bb, f16* __restrict__ out,
                               int B, int Cin, int T, int H, int W, int Kreal, int Kpad) {
  __shared__ int koff[896];  // per-k element offset
  __shared__ int kchk[896];  // per-k packed (dt,dy,dx)+8 for edge tiles
  __shared__ f16 bsh[(NT == 2) ? (32 * 448) : 1];
  const int HW = H * W, THW = T * HW;
  for (int k = threadIdx.x; k < Kpad; k += blockDim.x) {
    int doff = 0, pk = 0;
    if (k < Kreal) {
      int cin = k / 27, r = k % 27;
      int dt = r / 9 - 1, dy = (r / 3) % 3 - 1, dx = r % 3 - 1;
      doff = cin * THW + dt * HW + dy * W + dx;
      pk = (dt + 8) | ((dy + 8) << 8) | ((dx + 8) << 16);
    }
    koff[k] = doff;
    kchk[k] = pk;
  }
  if constexpr (NT == 2) {
    // TDM: 1-D tile of 14336 f16 (32x448) global -> LDS, tracked by TENSORcnt.
    if (threadIdx.x < 32) {
      unsigned lds = (unsigned)(size_t)&bsh[0];
      unsigned long long ga = (unsigned long long)(size_t)wt;
      v4i g0;
      g0[0] = 1;                                   // count=1, user descriptor
      g0[1] = (int)lds;                            // lds_addr
      g0[2] = (int)(ga & 0xffffffffu);             // global_addr[31:0]
      g0[3] = (int)((ga >> 32) & 0x01ffffffu) | (int)(2u << 30);  // addr hi | type=2
      v8i g1;
      g1[0] = 0x00010000;                          // data_size=1 (2 bytes)
      g1[1] = (int)(0x3800u << 16);                // tensor_dim0 lo16 (14336)
      g1[2] = (int)(1u << 16);                     // td0 hi16=0 | tensor_dim1=1
      g1[3] = (int)(0x3800u << 16);                // tile_dim0 = 14336
      g1[4] = 0;                                   // tile_dim1/2 unused (1-D)
      g1[5] = 14336;                               // tensor_dim0_stride
      g1[6] = 0;
      g1[7] = 0;
      asm volatile("tensor_load_to_lds %0, %1" :: "s"(g0), "s"(g1) : "memory");
      __builtin_amdgcn_s_wait_tensorcnt(0);
    }
  }
  __syncthreads();

  int lane = threadIdx.x & 31, half = lane >> 4, lm = lane & 15;
  int H2 = H >> 1, W2 = W >> 1, W4 = W2 >> 2;
  int wid = blockIdx.x * (blockDim.x >> 5) + (threadIdx.x >> 5);
  int w4 = wid % W4;  int t1 = wid / W4;
  int h2 = t1 % H2;   t1 /= H2;
  int t  = t1 % T;    int b = t1 / T;
  int pooled = lm >> 2, pos = lm & 3;
  int y = 2 * h2 + (pos >> 1);
  int x = 2 * (w4 * 4 + pooled) + (pos & 1);
  const f16* inb = in + (long)b * Cin * THW;
  int vox = t * HW + y * W + x;

  v8f zero = {};
  v8f acc[NT];
#pragma unroll
  for (int n = 0; n < NT; ++n) acc[n] = zero;

  auto bmm = [&](const v16h& a, int kb) {
#pragma unroll
    for (int n = 0; n < NT; ++n) {
      v16h bf;
      if constexpr (NT == 2)
        bf = *(const v16h*)(bsh + (n * 16 + lm) * 448 + kb + half * 16);  // ds_load
      else
        bf = *(const v16h*)(wt + (long)(n * 16 + lm) * Kpad + kb + half * 16);
      acc[n] = __builtin_amdgcn_wmma_f32_16x16x32_f16(false, a, false, bf,
                                                      (short)0, acc[n], false, false);
    }
  };

  bool edge = (t < 1) || (t > T - 2) || (y < 1) || (y > H - 2) ||
              (x < 1) || (x > W - 2);
  unsigned eb = wave_ballot(edge);
  const int fullsteps = Kreal >> 5;   // K-steps fully below Kreal

  if (eb == 0u) {
    // interior tile: offset-only gather, no bounds / K masks
    for (int s = 0; s < fullsteps; ++s) {
      int kb = s << 5;
      v16h a;
#pragma unroll
      for (int i = 0; i < 8; ++i) {
        int k0 = kb + a_kpos(i, half);
        int2 o2 = *(const int2*)(koff + k0);
        a[2 * i]     = inb[vox + o2.x];
        a[2 * i + 1] = inb[vox + o2.y];
      }
      bmm(a, kb);
    }
  } else {
    for (int s = 0; s < fullsteps; ++s) {
      int kb = s << 5;
      v16h a;
#pragma unroll
      for (int i = 0; i < 8; ++i) {
        int k0 = kb + a_kpos(i, half);
        int2 o2 = *(const int2*)(koff + k0);
        int2 c2 = *(const int2*)(kchk + k0);
#pragma unroll
        for (int j = 0; j < 2; ++j) {
          int ck = j ? c2.y : c2.x;
          int of = j ? o2.y : o2.x;
          int ti = t + ((ck & 0xff) - 8);
          int yi = y + (((ck >> 8) & 0xff) - 8);
          int xi = x + (((ck >> 16) & 0xff) - 8);
          f16 v = (f16)0.0f;
          if ((unsigned)ti < (unsigned)T && (unsigned)yi < (unsigned)H &&
              (unsigned)xi < (unsigned)W)
            v = inb[vox + of];
          a[2 * i + j] = v;
        }
      }
      bmm(a, kb);
    }
  }
  if (Kreal & 31) {  // masked tail step (stem1: K 416..431; stem2: none)
    int kb = fullsteps << 5;
    v16h a;
#pragma unroll
    for (int i = 0; i < 8; ++i) {
      int k0 = kb + a_kpos(i, half);
#pragma unroll
      for (int j = 0; j < 2; ++j) {
        int k = k0 + j;
        f16 v = (f16)0.0f;
        if (k < Kreal) {
          int ck = kchk[k];
          int ti = t + ((ck & 0xff) - 8);
          int yi = y + (((ck >> 8) & 0xff) - 8);
          int xi = x + (((ck >> 16) & 0xff) - 8);
          if ((unsigned)ti < (unsigned)T && (unsigned)yi < (unsigned)H &&
              (unsigned)xi < (unsigned)W)
            v = inb[vox + koff[k]];
        }
        a[2 * i + j] = v;
      }
    }
    bmm(a, kb);
  }

  // Epilogue: D lane holds (M = r+8*half, Nch = lane&15). Pool groups of 4 rows.
  const int Cout = NT * 16;
#pragma unroll
  for (int n = 0; n < NT; ++n) {
    int co = n * 16 + lm;
    float bias = cb[co], gg = bg[co], bbe = bb[co];
    float p0 = 0.0f, p1 = 0.0f;
#pragma unroll
    for (int r = 0; r < 4; ++r) {
      p0 = fmaxf(p0, fmaxf(gg * (acc[n][r] + bias) + bbe, 0.0f));
      p1 = fmaxf(p1, fmaxf(gg * (acc[n][4 + r] + bias) + bbe, 0.0f));
    }
    long ob = (((long)b * Cout + co) * T + t) * (H2 * W2) + h2 * W2 + w4 * 4;
    out[ob + 2 * half]     = (f16)p0;
    out[ob + 2 * half + 1] = (f16)p1;
  }
}

// ---------------- patch embed 4x4x4 stride 4 + bias + LIF -> tokens ---------
__global__ void patch_lif_wmma(const f16* __restrict__ in, const f16* __restrict__ wt,
                               const float* __restrict__ cb, float* __restrict__ xtok) {
  int lane = threadIdx.x & 31, half = lane >> 4, lm = lane & 15;
  int wid = blockIdx.x * (blockDim.x >> 5) + (threadIdx.x >> 5);
  int mt = wid >> 2, nt = wid & 3;
  int m = mt * 16 + lm;
  int b = (m >= 640) ? 1 : 0;           // m = b*640 + tp*16 + yp*4 + xp
  int mm = m - b * 640;
  int tp = mm >> 4, yp = (mm >> 2) & 3, xp = mm & 3;
  const f16* brow = wt + (long)(nt * 16 + lm) * 4096;
  v8f acc = {};
  for (int kb = 0; kb < 4096; kb += 32) {
    v16h a;
#pragma unroll
    for (int i = 0; i < 8; ++i) {
      int k0 = kb + a_kpos(i, half);
#pragma unroll
      for (int j = 0; j < 2; ++j) {
        int k = k0 + j;
        int cin = k >> 6, r = k & 63;
        int kd = r >> 4, kh = (r >> 2) & 3, kw = r & 3;
        a[2 * i + j] = in[(((long)b * 64 + cin) * 160 + tp * 4 + kd) * 256 +
                          (yp * 4 + kh) * 16 + (xp * 4 + kw)];
      }
    }
    v16h bf = *(const v16h*)(brow + kb + half * 16);
    __builtin_prefetch(brow + kb + 64, 0, 0);
    acc = __builtin_amdgcn_wmma_f32_16x16x32_f16(false, a, false, bf, (short)0,
                                                 acc, false, false);
  }
  int co = nt * 16 + lm;
  float bs = cb[co];
#pragma unroll
  for (int r = 0; r < 8; ++r) {
    int mr = mt * 16 + r + 8 * half;
    int b2 = (mr >= 640) ? 1 : 0;
    int mm2 = mr - b2 * 640;
    int tp2 = mm2 >> 4, yp2 = (mm2 >> 2) & 3, xp2 = mm2 & 3;
    float v = acc[r] + bs;
    float spike = (v * 0.5f >= 1.0f) ? 1.0f : 0.0f;  // LIF: v/tau >= vth
    xtok[(((long)tp2 * 2 + b2) * 16 + (yp2 * 4 + xp2)) * 64 + co] = spike;
  }
}

// ------------------------------ token GEMM ----------------------------------
// A: M x K f16 row-major; Wt: [N][K] f16. Modes: 0=bias, 1=bias+residual,
// 2=bias+LIF->f16, 3=bias+LIF+residual->f32.
__global__ void gemm_wmma(const f16* __restrict__ A, const f16* __restrict__ Wt,
                          const float* __restrict__ bias, const float* __restrict__ resid,
                          float* __restrict__ of32, f16* __restrict__ of16,
                          int M, int N, int K, int mode) {
  int lane = threadIdx.x & 31, half = lane >> 4, lm = lane & 15;
  int wid = blockIdx.x * (blockDim.x >> 5) + (threadIdx.x >> 5);
  int ntiles = N >> 4;
  int mt = wid / ntiles, nt = wid % ntiles;
  if (mt * 16 >= M) return;
  const f16* arow = A + (long)(mt * 16 + lm) * K;
  const f16* brow = Wt + (long)(nt * 16 + lm) * K;
  v8f acc = {};
  for (int kb = 0; kb < K; kb += 32) {
    v16h a = load_a_frag(arow, kb, half);
    v16h bf = *(const v16h*)(brow + kb + half * 16);
    __builtin_prefetch(arow + kb + 32, 0, 0);
    acc = __builtin_amdgcn_wmma_f32_16x16x32_f16(false, a, false, bf, (short)0,
                                                 acc, false, false);
  }
  int col = nt * 16 + lm;
  float bs = bias ? bias[col] : 0.0f;
#pragma unroll
  for (int r = 0; r < 8; ++r) {
    int row = mt * 16 + r + 8 * half;
    long o = (long)row * N + col;
    float v = acc[r] + bs;
    if (mode == 0)      of32[o] = v;
    else if (mode == 1) of32[o] = resid[o] + v;
    else if (mode == 2) of16[o] = (f16)((v * 0.5f >= 1.0f) ? 1.0f : 0.0f);
    else                of32[o] = resid[o] + ((v * 0.5f >= 1.0f) ? 1.0f : 0.0f);
  }
}

// ------------------------------ layernorm -----------------------------------
__global__ void layernorm16(const float* __restrict__ x, const float* __restrict__ g,
                            const float* __restrict__ b, f16* __restrict__ out, int ntok) {
  int t = blockIdx.x * blockDim.x + threadIdx.x;
  if (t >= ntok) return;
  const float* p = x + (long)t * 64;
  float m = 0.0f;
  for (int c = 0; c < 64; ++c) m += p[c];
  m *= (1.0f / 64.0f);
  float v = 0.0f;
  for (int c = 0; c < 64; ++c) { float d = p[c] - m; v += d * d; }
  v *= (1.0f / 64.0f);
  float inv = rsqrtf(v + 1e-5f);
  f16* o = out + (long)t * 64;
  for (int c = 0; c < 64; ++c) o[c] = (f16)((p[c] - m) * inv * g[c] + b[c]);
}

// -------------------------- BRA routing pieces ------------------------------
// token linear index = (t*2 + b)*16 + n ; region r = wt*8+wn, token i = a*2+c2
__global__ void region_mean(const float* __restrict__ qkv, float* __restrict__ qm,
                            float* __restrict__ km) {
  int id = blockIdx.x * blockDim.x + threadIdx.x;
  if (id >= 2 * 2 * 64 * 64) return;
  int c = id & 63, r = (id >> 6) & 63, b = (id >> 12) & 1, isk = id >> 13;
  int wt = r >> 3, wn = r & 7;
  float s = 0.0f;
  for (int a = 0; a < 5; ++a)
    for (int c2 = 0; c2 < 2; ++c2) {
      int t = wt * 5 + a, n = wn * 2 + c2;
      s += qkv[((long)(t * 2 + b) * 16 + n) * 192 + isk * 64 + c];
    }
  (isk ? km : qm)[((b * 64) + r) * 64 + c] = s * 0.1f;
}

__global__ void region_aff(const float* __restrict__ qm, const float* __restrict__ km,
                           float* __restrict__ aff) {
  int id = blockIdx.x * blockDim.x + threadIdx.x;
  if (id >= 2 * 64 * 64) return;
  int s = id & 63, r = (id >> 6) & 63, b = id >> 12;
  const float* q = qm + (b * 64 + r) * 64;
  const float* k = km + (b * 64 + s) * 64;
  float a = 0.0f;
  for (int c = 0; c < 64; ++c) a += q[c] * k[c];
  aff[id] = a;
}

__global__ void topk4(const float* __restrict__ aff, int* __restrict__ idx) {
  int id = blockIdx.x * blockDim.x + threadIdx.x;
  if (id >= 128) return;
  const float* row = aff + id * 64;
  unsigned long long used = 0ull;
  for (int j = 0; j < 4; ++j) {
    float best = -1e30f; int bi = 0;
    for (int s = 0; s < 64; ++s) {
      if ((used >> s) & 1ull) continue;
      float v = row[s];
      if (v > best) { best = v; bi = s; }
    }
    used |= 1ull << bi;
    idx[id * 4 + j] = bi;
  }
}

// grid = (b*64+r) over 128 blocks; block = 64 threads: h = tid>>4, tq = tid&15
__global__ void bra_attn(const float* __restrict__ qkv, const int* __restrict__ idx,
                         float* __restrict__ obuf) {
  int tq = threadIdx.x & 15;
  int h  = threadIdx.x >> 4;
  if (tq >= 10) return;
  int r = blockIdx.x & 63, b = blockIdx.x >> 6;
  int wt = r >> 3, wn = r & 7;
  int t = wt * 5 + (tq >> 1), n = wn * 2 + (tq & 1);
  long qb = ((long)(t * 2 + b) * 16 + n) * 192 + h * 16;
  float qv[16];
#pragma unroll
  for (int c = 0; c < 16; ++c) qv[c] = qkv[qb + c];
  float sc[40];
  float mx = -1e30f;
#pragma unroll
  for (int j = 0; j < 4; ++j) {
    int rs = idx[(b * 64 + r) * 4 + j];
    int wts = rs >> 3, wns = rs & 7;
#pragma unroll
    for (int i = 0; i < 10; ++i) {
      int ts = wts * 5 + (i >> 1), ns = wns * 2 + (i & 1);
      long kb = ((long)(ts * 2 + b) * 16 + ns) * 192 + 64 + h * 16;
      float d = 0.0f;
#pragma unroll
      for (int c = 0; c < 16; ++c) d += qv[c] * qkv[kb + c];
      d *= 0.25f;  // hd^-0.5, hd=16
      sc[j * 10 + i] = d;
      mx = fmaxf(mx, d);
    }
  }
  float den = 0.0f;
#pragma unroll
  for (int s = 0; s < 40; ++s) { sc[s] = __expf(sc[s] - mx); den += sc[s]; }
  float inv = 1.0f / den;
  float ov[16];
#pragma unroll
  for (int c = 0; c < 16; ++c) ov[c] = 0.0f;
#pragma unroll
  for (int j = 0; j < 4; ++j) {
    int rs = idx[(b * 64 + r) * 4 + j];
    int wts = rs >> 3, wns = rs & 7;
#pragma unroll
    for (int i = 0; i < 10; ++i) {
      int ts = wts * 5 + (i >> 1), ns = wns * 2 + (i & 1);
      long vb = ((long)(ts * 2 + b) * 16 + ns) * 192 + 128 + h * 16;
      float w = sc[j * 10 + i] * inv;
#pragma unroll
      for (int c = 0; c < 16; ++c) ov[c] += w * qkv[vb + c];
    }
  }
  long ob = ((long)(t * 2 + b) * 16 + n) * 64 + h * 16;
#pragma unroll
  for (int c = 0; c < 16; ++c) obuf[ob + c] = ov[c];
}

// depthwise 5x5 lepe on v over (T=40, N=16) grid, added to attention output
__global__ void lepe_add(const float* __restrict__ qkv, const float* __restrict__ lw,
                         const float* __restrict__ obuf, f16* __restrict__ olep) {
  int id = blockIdx.x * blockDim.x + threadIdx.x;
  if (id >= 81920) return;
  int c = id & 63, tok = id >> 6;
  int n = tok & 15, b = (tok >> 4) & 1, t = tok >> 5;
  float s = 0.0f;
#pragma unroll
  for (int dt = -2; dt <= 2; ++dt)
#pragma unroll
    for (int dn = -2; dn <= 2; ++dn) {
      int ti = t + dt, ni = n + dn;
      if ((unsigned)ti < 40u && (unsigned)ni < 16u)
        s += qkv[((long)(ti * 2 + b) * 16 + ni) * 192 + 128 + c] *
             lw[c * 25 + (dt + 2) * 5 + (dn + 2)];
    }
  olep[id] = (f16)(obuf[id] + s);
}

// --------------------------- decoder / head ---------------------------------
// grid = (b*64+c) over 128 blocks; block = 640 threads: t = tid>>4, n = tid&15
__global__ void tok2img(const float* __restrict__ x, float* __restrict__ y0) {
  int t = threadIdx.x >> 4, n = threadIdx.x & 15;
  int b = blockIdx.x >> 6, c = blockIdx.x & 63;
  long o = ((long)blockIdx.x * 40 + t) * 16 + n;
  y0[o] = x[((long)(t * 2 + b) * 16 + n) * 64 + c];
}

// nearest x2 temporal repeat + conv(3,1,1) pad1 + affine + ELU
__global__ void up_conv_elu(const float* __restrict__ in, const float* __restrict__ w,
                            const float* __restrict__ cb, const float* __restrict__ g,
                            const float* __restrict__ be, float* __restrict__ out,
                            int B, int Cin, int Cout, int Tout) {
  int id = blockIdx.x * blockDim.x + threadIdx.x;
  int total = B * Cout * Tout * 16;
  if (id >= total) return;
  int sp = id & 15, t = (id >> 4) % Tout;
  int co = (id / (16 * Tout)) % Cout;
  int b = id / (16 * Tout * Cout);
  int Tin = Tout >> 1;
  float s = 0.0f;
  for (int kt = 0; kt < 3; ++kt) {
    int tt = t + kt - 1;
    if ((unsigned)tt < (unsigned)Tout) {
      int ti = tt >> 1;
      const float* ip = in + (((long)b * Cin) * Tin + ti) * 16 + sp;
      const float* wp = w + (co * Cin) * 3 + kt;
      for (int ci = 0; ci < Cin; ++ci) s += ip[(long)ci * Tin * 16] * wp[ci * 3];
    }
  }
  float v = g[co] * (s + cb[co]) + be[co];
  out[id] = v > 0.0f ? v : (__expf(v) - 1.0f);
}

__global__ void final_head(const float* __restrict__ u2, const float* __restrict__ lw,
                           const float* __restrict__ lb, float* __restrict__ out) {
  int id = blockIdx.x * blockDim.x + threadIdx.x;
  if (id >= 320) return;
  int t = id % 160, b = id / 160;
  float acc = 0.0f;
  for (int c = 0; c < 32; ++c) {
    const float* p = u2 + (((long)b * 32 + c) * 160 + t) * 16;
    float s = 0.0f;
    for (int sp = 0; sp < 16; ++sp) s += p[sp];
    acc += lw[c] * s;
  }
  out[id] = acc + lb[0];
}

// ------------------------------- launcher -----------------------------------
extern "C" void kernel_launch(void* const* d_in, const int* in_sizes, int n_in,
                              void* d_out, int out_size, void* d_ws, size_t ws_size,
                              hipStream_t stream) {
  (void)in_sizes; (void)n_in; (void)out_size; (void)ws_size;
  auto F = [&](int i) { return (const float*)d_in[i]; };
  char* base = (char*)d_ws;
  size_t off = 0;
  auto alloc = [&](size_t bytes) -> void* {
    void* p = base + off;
    off = (off + bytes + 255) & ~(size_t)255;
    return p;
  };
  auto nb = [](long n) { return (int)((n + 255) / 256); };

  f16* wts1 = (f16*)alloc((size_t)32 * 448 * 2);
  f16* wts2 = (f16*)alloc((size_t)64 * 896 * 2);
  f16* wtp  = (f16*)alloc((size_t)64 * 4096 * 2);
  f16 *wtqkv[3], *wtwo[3], *wtfc1[3], *wtfc2[3];
  for (int i = 0; i < 3; ++i) {
    wtqkv[i] = (f16*)alloc((size_t)192 * 64 * 2);
    wtwo[i]  = (f16*)alloc((size_t)64 * 64 * 2);
    wtfc1[i] = (f16*)alloc((size_t)256 * 64 * 2);
    wtfc2[i] = (f16*)alloc((size_t)64 * 256 * 2);
  }
  f16* a0 = (f16*)alloc((size_t)2 * 16 * 160 * 64 * 64 * 2);
  f16* a1 = (f16*)alloc((size_t)2 * 32 * 160 * 32 * 32 * 2);
  f16* a2 = (f16*)alloc((size_t)2 * 64 * 160 * 16 * 16 * 2);
  float* xa  = (float*)alloc((size_t)81920 * 4);
  float* xb  = (float*)alloc((size_t)81920 * 4);
  f16*   lnb = (f16*)alloc((size_t)81920 * 2);
  float* qkvb = (float*)alloc((size_t)1280 * 192 * 4);
  float* qm   = (float*)alloc((size_t)2 * 64 * 64 * 4);
  float* km   = (float*)alloc((size_t)2 * 64 * 64 * 4);
  float* affb = (float*)alloc((size_t)2 * 64 * 64 * 4);
  int*   idxb = (int*)alloc((size_t)2 * 64 * 4 * 4);
  float* obuf = (float*)alloc((size_t)81920 * 4);
  f16*   olep = (f16*)alloc((size_t)81920 * 2);
  f16*   h1   = (f16*)alloc((size_t)1280 * 256 * 2);
  float* y0   = (float*)alloc((size_t)81920 * 4);
  float* u1   = (float*)alloc((size_t)2 * 64 * 80 * 16 * 4);
  float* u2   = (float*)alloc((size_t)2 * 32 * 160 * 16 * 4);

  // ---- pack weights to f16 [N][Kpad] ----
  pack_pad<<<nb(32L * 448), 256, 0, stream>>>(F(5), wts1, 432, 448, 32 * 448);
  pack_pad<<<nb(64L * 896), 256, 0, stream>>>(F(9), wts2, 864, 896, 64 * 896);
  pack_pad<<<nb(64L * 4096), 256, 0, stream>>>(F(13), wtp, 4096, 4096, 64 * 4096);
  for (int blk = 0; blk < 3; ++blk) {
    int pb = 15 + blk * 13;
    pack_T<<<nb(192L * 64), 256, 0, stream>>>(F(pb + 2), wtqkv[blk], 64, 192, 64, 192 * 64);
    pack_T<<<nb(64L * 64), 256, 0, stream>>>(F(pb + 5), wtwo[blk], 64, 64, 64, 64 * 64);
    pack_T<<<nb(256L * 64), 256, 0, stream>>>(F(pb + 9), wtfc1[blk], 64, 256, 64, 256 * 64);
    pack_T<<<nb(64L * 256), 256, 0, stream>>>(F(pb + 11), wtfc2[blk], 256, 64, 256, 64 * 256);
  }

  // ---- stem + patch ----
  stem0_fused<<<81920, 256, 0, stream>>>(F(0), F(1), F(2), F(3), F(4), a0);
  conv_pool_wmma<2><<<10240, 256, 0, stream>>>(a0, wts1, F(6), F(7), F(8), a1,
                                               2, 16, 160, 64, 64, 432, 448);
  conv_pool_wmma<4><<<2560, 256, 0, stream>>>(a1, wts2, F(10), F(11), F(12), a2,
                                              2, 32, 160, 32, 32, 864, 896);
  patch_lif_wmma<<<40, 256, 0, stream>>>(a2, wtp, F(14), xa);

  // ---- transformer blocks (x lives in xa across blocks) ----
  for (int blk = 0; blk < 3; ++blk) {
    int pb = 15 + blk * 13;
    layernorm16<<<5, 256, 0, stream>>>(xa, F(pb + 0), F(pb + 1), lnb, 1280);
    gemm_wmma<<<120, 256, 0, stream>>>(lnb, wtqkv[blk], F(pb + 3), nullptr,
                                       qkvb, nullptr, 1280, 192, 64, 0);
    region_mean<<<64, 256, 0, stream>>>(qkvb, qm, km);
    region_aff<<<32, 256, 0, stream>>>(qm, km, affb);
    topk4<<<1, 128, 0, stream>>>(affb, idxb);
    bra_attn<<<128, 64, 0, stream>>>(qkvb, idxb, obuf);
    lepe_add<<<320, 256, 0, stream>>>(qkvb, F(pb + 4), obuf, olep);
    gemm_wmma<<<40, 256, 0, stream>>>(olep, wtwo[blk], F(pb + 6), xa,
                                      xb, nullptr, 1280, 64, 64, 1);   // x_mid = x + attn
    layernorm16<<<5, 256, 0, stream>>>(xb, F(pb + 7), F(pb + 8), lnb, 1280);
    gemm_wmma<<<160, 256, 0, stream>>>(lnb, wtfc1[blk], F(pb + 10), nullptr,
                                       nullptr, h1, 1280, 256, 64, 2); // lif(fc1)
    gemm_wmma<<<40, 256, 0, stream>>>(h1, wtfc2[blk], F(pb + 12), xb,
                                      xa, nullptr, 1280, 64, 256, 3);  // x = x_mid + lif(fc2)
  }

  // ---- decoder + head ----
  tok2img<<<128, 640, 0, stream>>>(xa, y0);
  up_conv_elu<<<640, 256, 0, stream>>>(y0, F(54), F(55), F(56), F(57), u1, 2, 64, 64, 80);
  up_conv_elu<<<640, 256, 0, stream>>>(u1, F(58), F(59), F(60), F(61), u2, 2, 64, 32, 160);
  final_head<<<2, 160, 0, stream>>>(u2, F(62), F(63), (float*)d_out);
}
